// VishwamaiAttention_3728031613120
// MI455X (gfx1250) — matrix operand order repaired and checked by reading, not compiled, not run on hardware
//
#include <hip/hip_runtime.h>
#include <hip/hip_bf16.h>
#include <math.h>

// ---------------- problem constants ----------------
constexpr int SEQ   = 2048;
constexpr int HIDN  = 4096;
constexpr int NH    = 32;     // query heads
constexpr int NKV   = 8;      // kv heads
constexpr int HD    = 128;    // head dim
constexpr int QKVN  = (NH + 2 * NKV) * HD;   // 6144
constexpr int WINDOW = 1024;
constexpr float SOFTCAP = 50.0f;
constexpr float SCALE   = 0.08838834764831845f;   // 128^-0.5
constexpr float LN_ROPE_BASE = 9.210340371976184f; // ln(10000)

// ---------------- WMMA vector types ----------------
typedef __attribute__((ext_vector_type(16))) __bf16 v16bf;
typedef __attribute__((ext_vector_type(8)))  __bf16 v8bf;
typedef __attribute__((ext_vector_type(8)))  float  v8f;

// ---------------- TDM descriptor types ----------------
typedef __attribute__((ext_vector_type(4))) unsigned int u32x4;
typedef __attribute__((ext_vector_type(8))) int          i32x8;
typedef __attribute__((ext_vector_type(4))) int          i32x4;

// A-matrix (16x32 bf16) fragment, contiguous k order (matches B):
//   elements 0..7  -> K = kbase + half*16 + (0..7)
//   elements 8..15 -> K = kbase + half*16 + 8 + (0..7)
__device__ __forceinline__ v16bf load_a_frag(const __bf16* base, int hf) {
    union { v16bf v; v8bf h[2]; } u;
    u.h[0] = *(const v8bf*)(base + hf * 16);
    u.h[1] = *(const v8bf*)(base + hf * 16 + 8);
    return u.v;
}

__device__ __forceinline__ v8f wmma_bf16(v16bf a, v16bf b, v8f c) {
    return __builtin_amdgcn_wmma_f32_16x16x32_bf16(false, a, false, b,
                                                   (short)0, c, false, false);
}

__device__ __forceinline__ v8f vzero8() {
    return v8f{0.f, 0.f, 0.f, 0.f, 0.f, 0.f, 0.f, 0.f};
}

// Issue a 2D TDM tile load (bf16 elements, tile == tensor extent) into LDS.
//   lds_off  : LDS byte address of destination
//   gptr     : global address of tile origin
//   tile0    : contiguous row length (elements)
//   tile1    : number of rows
//   stride0  : row stride in elements
// Builds a D# per CDNA5 ISA 8.3/8.4: group0 = {count/flags, lds_addr, global
// addr[56:0], type=2}; group1 = {data_size=2B, tensor_dim0/1, tile_dim0/1,
// tensor_dim0_stride}. Groups 2/3 zero (2D tensor).
__device__ __forceinline__ void tdm_load_2d(unsigned lds_off, const void* gptr,
                                            unsigned tile0, unsigned tile1,
                                            unsigned stride0) {
    unsigned long long ga = (unsigned long long)(uintptr_t)gptr;
    u32x4 g0;
    g0[0] = 1u;                                        // count=1 (valid user D#)
    g0[1] = lds_off;                                   // LDS byte address
    g0[2] = (unsigned)(ga & 0xffffffffu);              // global addr [31:0]
    g0[3] = (unsigned)((ga >> 32) & 0x1ffffffu)        // global addr [56:32]
          | (2u << 30);                                // type = 2 ("image")
    i32x8 g1;
    g1[0] = (int)(1u << 16);                           // wg_mask=0, data_size=1 (2B)
    g1[1] = (int)((tile0 & 0xffffu) << 16);            // tensor_dim0[15:0]
    g1[2] = (int)((tile0 >> 16) | ((tile1 & 0xffffu) << 16)); // td0 hi | td1 lo
    g1[3] = (int)((tile1 >> 16) | ((tile0 & 0xffffu) << 16)); // td1 hi | tile_dim0
    g1[4] = (int)(tile1 & 0xffffu);                    // tile_dim1 | tile_dim2=0
    g1[5] = (int)stride0;                              // tensor_dim0_stride[31:0]
    g1[6] = 0;                                         // stride0 hi | stride1 lo
    g1[7] = 0;                                         // stride1 hi
    i32x4 gz = {0, 0, 0, 0};
#if __clang_major__ >= 23
    i32x8 gz8 = {0, 0, 0, 0, 0, 0, 0, 0};
    __builtin_amdgcn_tensor_load_to_lds(g0, g1, gz, gz, gz8, 0);
#else
    __builtin_amdgcn_tensor_load_to_lds(g0, g1, gz, gz, 0);
#endif
}

// =====================================================================
// Kernel 1: QKV projection (bf16 WMMA, f32 acc) + RoPE + q/k/v scatter
// grid: (QKVN/128, SEQ/128), block: 256
// =====================================================================
__global__ __launch_bounds__(256)
void qkv_rope_kernel(const float* __restrict__ X,
                     const int*   __restrict__ pos_idx,
                     const float* __restrict__ W,
                     __bf16* __restrict__ wq,
                     __bf16* __restrict__ wk,
                     __bf16* __restrict__ wv) {
    __shared__ __align__(32) float smem[128 * 128];          // 64 KB, reused
    __bf16* lA  = (__bf16*)smem;                              // [128][32]
    __bf16* lBt = (__bf16*)smem + 128 * 32;                   // [128 n][32 k]

    const int tid  = threadIdx.x;
    const int wave = tid >> 5, lane = tid & 31;
    const int hf = lane >> 4, l16 = lane & 15;
    const int waveM = wave & 3, waveN = wave >> 2;
    const int rowBase = blockIdx.y * 128;
    const int colBase = blockIdx.x * 128;

    v8f acc[2][4];
#pragma unroll
    for (int mt = 0; mt < 2; mt++)
#pragma unroll
        for (int nt = 0; nt < 4; nt++) acc[mt][nt] = vzero8();

    for (int k0 = 0; k0 < HIDN; k0 += 32) {
        for (int idx = tid; idx < 128 * 32; idx += 256) {
            int r = idx >> 5, c = idx & 31;
            lA[idx] = (__bf16)X[(size_t)(rowBase + r) * HIDN + k0 + c];
        }
        for (int idx = tid; idx < 128 * 32; idx += 256) {
            int n = idx & 127, k = idx >> 7;                  // coalesced over n
            lBt[n * 32 + k] = (__bf16)W[(size_t)(k0 + k) * QKVN + colBase + n];
        }
        __syncthreads();
#pragma unroll
        for (int mt = 0; mt < 2; mt++) {
            v16bf a = load_a_frag(lA + (waveM * 32 + mt * 16 + l16) * 32, hf);
#pragma unroll
            for (int nt = 0; nt < 4; nt++) {
                v16bf b = *(const v16bf*)(lBt + (waveN * 64 + nt * 16 + l16) * 32 + hf * 16);
                acc[mt][nt] = wmma_bf16(a, b, acc[mt][nt]);
            }
        }
        __syncthreads();
    }

    // dump accumulators (C layout: row = r + half*8, col = l16) to LDS f32 tile
#pragma unroll
    for (int mt = 0; mt < 2; mt++)
#pragma unroll
        for (int nt = 0; nt < 4; nt++)
#pragma unroll
            for (int r = 0; r < 8; r++) {
                int m = waveM * 32 + mt * 16 + r + hf * 8;
                int n = waveN * 64 + nt * 16 + l16;
                smem[m * 128 + n] = acc[mt][nt][r];
            }
    __syncthreads();

    const bool isQ = colBase < NH * HD;
    const bool isK = (colBase >= NH * HD) && (colBase < (NH + NKV) * HD);
    for (int idx = tid; idx < 128 * 128; idx += 256) {
        int m = idx >> 7, dd = idx & 127;
        int row = rowBase + m;
        float val = smem[idx];
        int pos = pos_idx[row];
        if (isQ || isK) {
            float partner = smem[idx ^ 64];                   // dd ^ 64, same row
            int f = dd & 63;
            float inv = __expf(-(2.0f * (float)f / 128.0f) * LN_ROPE_BASE);
            float ph = (float)pos * inv;
            float sn, cs;
            __sincosf(ph, &sn, &cs);
            val = (dd < 64) ? (val * cs - partner * sn)
                            : (val * cs + partner * sn);
        }
        if (isQ) {
            int hq = colBase >> 7;
            wq[(size_t)row * (NH * HD) + hq * HD + dd] = (__bf16)val;
        } else if (isK) {
            int hk = (colBase - NH * HD) >> 7;
            wk[(size_t)pos * (NKV * HD) + hk * HD + dd] = (__bf16)val;
        } else {
            int hv = (colBase - (NH + NKV) * HD) >> 7;
            wv[(size_t)pos * (NKV * HD) + hv * HD + dd] = (__bf16)val;
        }
    }
}

// =====================================================================
// Kernel 2: windowed-causal GQA flash attention with tanh softcap
// grid: (NH, SEQ/128), block: 256 (8 waves x 16 query rows)
// Q and K tiles staged by the Tensor Data Mover; V staged transposed.
// =====================================================================
__global__ __launch_bounds__(256)
void attn_kernel(const __bf16* __restrict__ wq,
                 const __bf16* __restrict__ wk,
                 const __bf16* __restrict__ wv,
                 __bf16* __restrict__ wo) {
    __shared__ __align__(32) __bf16 lQ[128 * 128];   // 32 KB [row][d]
    __shared__ __align__(32) __bf16 lK[32 * 128];    //  8 KB [key][d]
    __shared__ __align__(32) __bf16 lVt[128 * 32];   //  8 KB [d][key]
    __shared__ __align__(32) __bf16 lP[8][16 * 32];  //  8 KB per-wave [row][key]

    const int tid  = threadIdx.x;
    const int wave = tid >> 5, lane = tid & 31;
    const int hf = lane >> 4, l16 = lane & 15;
    const int h = blockIdx.x;
    const int qbase = blockIdx.y * 128;
    const int kvh = h >> 2;                          // H / KV = 4

    const unsigned ldsQ = (unsigned)(uintptr_t)(void*)lQ;   // flat[31:0] = LDS off
    const unsigned ldsK = (unsigned)(uintptr_t)(void*)lK;

    // TDM: stage the block's 128 query rows for this head (128x128 bf16 tile)
    if (wave == 0)
        tdm_load_2d(ldsQ, wq + (size_t)qbase * (NH * HD) + h * HD,
                    /*tile0=*/HD, /*tile1=*/128, /*stride0=*/NH * HD);

    v8f oacc[8];
#pragma unroll
    for (int n = 0; n < 8; n++) oacc[n] = vzero8();
    float mrow[8], lrow[8];
#pragma unroll
    for (int r = 0; r < 8; r++) { mrow[r] = -INFINITY; lrow[r] = 0.0f; }

    int j0 = qbase - (WINDOW - 1);
    if (j0 < 0) j0 = 0;
    j0 &= ~31;
    const int jend = qbase + 128;

    for (int jb = j0; jb < jend; jb += 32) {
        // TDM: stage K tile (32 rows x 128 dims, key-major)
        if (wave == 0)
            tdm_load_2d(ldsK, wk + (size_t)jb * (NKV * HD) + kvh * HD,
                        /*tile0=*/HD, /*tile1=*/32, /*stride0=*/NKV * HD);

        // stage V tile transposed: lVt[d][key]  (uint4 = 8 bf16 global loads)
        {
            const uint4* gv = (const uint4*)wv;
            for (int i = tid; i < 32 * 16; i += 256) {
                int key = i >> 4, c8 = i & 15;
                uint4 val = gv[(size_t)(jb + key) * (NKV * HD / 8) + kvh * (HD / 8) + c8];
                const __bf16* e = (const __bf16*)&val;
#pragma unroll
                for (int q = 0; q < 8; q++)
                    lVt[(c8 * 8 + q) * 32 + key] = e[q];
            }
        }
        // drain the TDM before releasing readers through the barrier
        if (wave == 0) __builtin_amdgcn_s_wait_tensorcnt(0);
        __syncthreads();

        // ---- scores: (16 q rows) x (32 keys), K-dim = 128 head dims ----
        v8f sacc[2];
        sacc[0] = vzero8(); sacc[1] = vzero8();
#pragma unroll
        for (int kc = 0; kc < 4; kc++) {
            v16bf a = load_a_frag(lQ + (wave * 16 + l16) * 128 + kc * 32, hf);
#pragma unroll
            for (int t = 0; t < 2; t++) {
                v16bf b = *(const v16bf*)(lK + (t * 16 + l16) * 128 + kc * 32 + hf * 16);
                sacc[t] = wmma_bf16(a, b, sacc[t]);
            }
        }

        // ---- softcap + band/causal mask + online softmax ----
        float p[2][8], tmax[8], alpha[8];
#pragma unroll
        for (int r = 0; r < 8; r++) {
            int i = qbase + wave * 16 + r + hf * 8;
            float best = -INFINITY;
#pragma unroll
            for (int t = 0; t < 2; t++) {
                int j = jb + t * 16 + l16;
                float s;
                if (j > i || (i - j) > (WINDOW - 1)) s = -INFINITY;
                else s = SOFTCAP * tanhf(sacc[t][r] * (SCALE / SOFTCAP));
                p[t][r] = s;
                best = fmaxf(best, s);
            }
#pragma unroll
            for (int off = 1; off < 16; off <<= 1)
                best = fmaxf(best, __shfl_xor(best, off, 32));
            tmax[r] = best;
        }
#pragma unroll
        for (int r = 0; r < 8; r++) {
            float mnew = fmaxf(mrow[r], tmax[r]);
            bool dead = !(mnew > -INFINITY);
            alpha[r] = dead ? 0.0f : __expf(mrow[r] - mnew);
            float sum = 0.0f;
#pragma unroll
            for (int t = 0; t < 2; t++) {
                float pv = dead ? 0.0f : __expf(p[t][r] - mnew);
                p[t][r] = pv;
                sum += pv;
            }
#pragma unroll
            for (int off = 1; off < 16; off <<= 1)
                sum += __shfl_xor(sum, off, 32);
            lrow[r] = lrow[r] * alpha[r] + sum;
            mrow[r] = mnew;
        }
#pragma unroll
        for (int n = 0; n < 8; n++)
#pragma unroll
            for (int r = 0; r < 8; r++) oacc[n][r] *= alpha[r];

        // reshape P: C layout -> A layout via per-wave LDS (no barrier: same wave)
#pragma unroll
        for (int t = 0; t < 2; t++)
#pragma unroll
            for (int r = 0; r < 8; r++)
                lP[wave][(r + hf * 8) * 32 + t * 16 + l16] = (__bf16)p[t][r];

        // ---- O += P(16x32) @ V(32x128) ----
        v16bf aP = load_a_frag(&lP[wave][l16 * 32], hf);
#pragma unroll
        for (int n = 0; n < 8; n++) {
            v16bf b = *(const v16bf*)(lVt + (n * 16 + l16) * 32 + hf * 16);
            oacc[n] = wmma_bf16(aP, b, oacc[n]);
        }
        __syncthreads();
    }

    // finalize: divide by row sums, write bf16 attention output
#pragma unroll
    for (int n = 0; n < 8; n++)
#pragma unroll
        for (int r = 0; r < 8; r++) {
            int row = qbase + wave * 16 + r + hf * 8;
            int col = h * HD + n * 16 + l16;
            float d = lrow[r];
            float val = (d > 0.0f) ? oacc[n][r] / d : 0.0f;
            wo[(size_t)row * (NH * HD) + col] = (__bf16)val;
        }
}

// =====================================================================
// Kernel 3: output projection  out = attn(bf16) @ o_w(f32), f32 out
// grid: (HIDN/128, SEQ/128), block: 256
// =====================================================================
__global__ __launch_bounds__(256)
void oproj_kernel(const __bf16* __restrict__ A,
                  const float*  __restrict__ W,
                  float* __restrict__ out) {
    __shared__ __align__(32) __bf16 lA[128 * 32];
    __shared__ __align__(32) __bf16 lBt[128 * 32];

    const int tid  = threadIdx.x;
    const int wave = tid >> 5, lane = tid & 31;
    const int hf = lane >> 4, l16 = lane & 15;
    const int waveM = wave & 3, waveN = wave >> 2;
    const int rowBase = blockIdx.y * 128;
    const int colBase = blockIdx.x * 128;

    v8f acc[2][4];
#pragma unroll
    for (int mt = 0; mt < 2; mt++)
#pragma unroll
        for (int nt = 0; nt < 4; nt++) acc[mt][nt] = vzero8();

    for (int k0 = 0; k0 < HIDN; k0 += 32) {
        for (int idx = tid; idx < 128 * 32; idx += 256) {
            int r = idx >> 5, c = idx & 31;
            lA[idx] = A[(size_t)(rowBase + r) * HIDN + k0 + c];
        }
        for (int idx = tid; idx < 128 * 32; idx += 256) {
            int n = idx & 127, k = idx >> 7;
            lBt[n * 32 + k] = (__bf16)W[(size_t)(k0 + k) * HIDN + colBase + n];
        }
        __syncthreads();
#pragma unroll
        for (int mt = 0; mt < 2; mt++) {
            v16bf a = load_a_frag(lA + (waveM * 32 + mt * 16 + l16) * 32, hf);
#pragma unroll
            for (int nt = 0; nt < 4; nt++) {
                v16bf b = *(const v16bf*)(lBt + (waveN * 64 + nt * 16 + l16) * 32 + hf * 16);
                acc[mt][nt] = wmma_bf16(a, b, acc[mt][nt]);
            }
        }
        __syncthreads();
    }

#pragma unroll
    for (int mt = 0; mt < 2; mt++)
#pragma unroll
        for (int nt = 0; nt < 4; nt++)
#pragma unroll
            for (int r = 0; r < 8; r++) {
                int m = rowBase + waveM * 32 + mt * 16 + r + hf * 8;
                int n = colBase + waveN * 64 + nt * 16 + l16;
                out[(size_t)m * HIDN + n] = acc[mt][nt][r];
            }
}

// =====================================================================
// host launcher
// =====================================================================
extern "C" void kernel_launch(void* const* d_in, const int* in_sizes, int n_in,
                              void* d_out, int out_size, void* d_ws, size_t ws_size,
                              hipStream_t stream) {
    (void)in_sizes; (void)n_in; (void)out_size; (void)ws_size;
    const float* hidden = (const float*)d_in[0];
    const int*   posidx = (const int*)d_in[1];
    // d_in[2] k_cache, d_in[3] v_cache: reference writes full cache (identity
    // scatter) and never returns it; d_in[4] mask is the additive causal mask,
    // reproduced analytically inside attn_kernel. None need to be read.
    const float* qkv_w  = (const float*)d_in[5];
    const float* o_w    = (const float*)d_in[6];
    float* out = (float*)d_out;

    // workspace: bf16 intermediates (~40 MB total)
    __bf16* wq = (__bf16*)d_ws;                            // [SEQ][NH*HD]
    __bf16* wk = wq + (size_t)SEQ * NH * HD;               // [SEQ][NKV*HD]
    __bf16* wv = wk + (size_t)SEQ * NKV * HD;              // [SEQ][NKV*HD]
    __bf16* wo = wv + (size_t)SEQ * NKV * HD;              // [SEQ][NH*HD]

    qkv_rope_kernel<<<dim3(QKVN / 128, SEQ / 128), 256, 0, stream>>>(
        hidden, posidx, qkv_w, wq, wk, wv);
    attn_kernel<<<dim3(NH, SEQ / 128), 256, 0, stream>>>(wq, wk, wv, wo);
    oproj_kernel<<<dim3(HIDN / 128, SEQ / 128), 256, 0, stream>>>(wo, o_w, out);
}
